// SimpleGAT_49520972923232
// MI455X (gfx1250) — matrix operand order repaired and checked by reading, not compiled, not run on hardware
//
#include <hip/hip_runtime.h>
#include <math.h>

// ---------------------------------------------------------------------------
// Types for CDNA5 WMMA (wave32, 16x16x32 bf16 -> f32)
// ---------------------------------------------------------------------------
typedef __attribute__((ext_vector_type(16))) __bf16 v16bf;
typedef __attribute__((ext_vector_type(8)))  __bf16 v8bf;
typedef __attribute__((ext_vector_type(8)))  float  v8f;

#define N_NODES 20000
#define N_EDGES 320000

__device__ __forceinline__ unsigned short f2bf(float f) {
    unsigned int u = __builtin_bit_cast(unsigned int, f);
    unsigned int r = u + 0x7FFFu + ((u >> 16) & 1u);   // round-to-nearest-even
    return (unsigned short)(r >> 16);
}

__device__ __forceinline__ void atomicMaxF(float* addr, float v) {
    if (v >= 0.0f) atomicMax((int*)addr, __builtin_bit_cast(int, v));
    else           atomicMin((unsigned int*)addr, __builtin_bit_cast(unsigned int, v));
}

__device__ __forceinline__ float eluf(float v) {
    return v > 0.0f ? v : (__expf(v) - 1.0f);
}

// ---------------------------------------------------------------------------
// Generic fill
// ---------------------------------------------------------------------------
__global__ void fill_f32(float* __restrict__ p, float v, size_t n) {
    for (size_t i = blockIdx.x * (size_t)blockDim.x + threadIdx.x; i < n;
         i += (size_t)gridDim.x * blockDim.x)
        p[i] = v;
}

// fp32 -> bf16 bit-pattern cast (row-major)
__global__ void cast_bf16_k(const float* __restrict__ in, unsigned short* __restrict__ out, size_t n) {
    for (size_t i = blockIdx.x * (size_t)blockDim.x + threadIdx.x; i < n;
         i += (size_t)gridDim.x * blockDim.x)
        out[i] = f2bf(in[i]);
}

// ---------------------------------------------------------------------------
// Pack fp32 weight W[K x Nc] (row-major) into WMMA B-fragment layout (bf16).
// Layout: idx = ((kt*ntiles + nt)*32 + lane)*8 + v  holds bf16 pair
//   K = kt*32 + (lane>>4)*16 + 2v, 2v+1   at column  nt*16 + (lane&15)
// ---------------------------------------------------------------------------
__global__ void pack_b_wmma(const float* __restrict__ W, unsigned int* __restrict__ Bp,
                            int K, int Nc) {
    int ntiles = Nc >> 4, ktiles = K >> 5;
    int total = ktiles * ntiles * 256;
    int idx = blockIdx.x * blockDim.x + threadIdx.x;
    if (idx >= total) return;
    int v    = idx & 7;
    int lane = (idx >> 3) & 31;
    int tile = idx >> 8;                 // kt*ntiles + nt
    int nt = tile % ntiles;
    int kt = tile / ntiles;
    int n  = nt * 16 + (lane & 15);
    int k  = kt * 32 + (lane >> 4) * 16 + 2 * v;
    unsigned int lo = f2bf(W[(size_t)k * Nc + n]);
    unsigned int hi = f2bf(W[(size_t)(k + 1) * Nc + n]);
    Bp[idx] = lo | (hi << 16);
}

// ---------------------------------------------------------------------------
// WMMA GEMM: C[M x Nc] = A_bf16[M x K] @ Bpacked(bf16).  One wave computes a
// 16 x (NT*16) strip: the A fragment is loaded once per K-step and reused for
// NT v_wmma_f32_16x16x32_bf16 issues (4x less A traffic, 4 WMMA per 2 A-loads)
// ---------------------------------------------------------------------------
template <int NT>
__global__ void gemm_wmma_bf16(const unsigned short* __restrict__ A,
                               const unsigned int* __restrict__ Bp,
                               float* __restrict__ C,
                               int K, int Nc, int totalWaves) {
    int gw = (int)((blockIdx.x * (size_t)blockDim.x + threadIdx.x) >> 5);
    if (gw >= totalWaves) return;                 // whole-wave uniform exit
    int lane   = threadIdx.x & 31;
    int ntiles = Nc >> 4;
    int ngroups = ntiles / NT;
    int mt  = gw / ngroups;
    int ng  = gw - mt * ngroups;
    int nt0 = ng * NT;
    int mrow  = lane & 15;
    int khalf = lane >> 4;

    v8f acc[NT];
#pragma unroll
    for (int j = 0; j < NT; ++j)
        acc[j] = (v8f){0.f, 0.f, 0.f, 0.f, 0.f, 0.f, 0.f, 0.f};

    const unsigned short* arow = A + (size_t)(mt * 16 + mrow) * K + khalf * 16;
    const unsigned int*   bp   = Bp + ((size_t)nt0 * 32 + lane) * 8;
    size_t bstride = (size_t)ntiles * 256;        // dwords per K-tile of packed B

    for (int kt = 0; kt < K; kt += 32) {
        v8bf alo = *(const v8bf*)(arow);          // K = khalf*16 + 0..7
        v8bf ahi = *(const v8bf*)(arow + 8);      // K = khalf*16 + 8..15
        v16bf a = __builtin_shufflevector(alo, ahi,
                    0,1,2,3,4,5,6,7,8,9,10,11,12,13,14,15);
#pragma unroll
        for (int j = 0; j < NT; ++j) {
            v16bf b = *(const v16bf*)(bp + (size_t)j * 256);  // 32B fragment
            acc[j] = __builtin_amdgcn_wmma_f32_16x16x32_bf16(
                         false, a, false, b, (short)0, acc[j], false, false);
        }
        arow += 32;
        bp   += bstride;
    }

#pragma unroll
    for (int j = 0; j < NT; ++j) {
        int col = (nt0 + j) * 16 + mrow;
#pragma unroll
        for (int r = 0; r < 8; ++r)
            C[(size_t)(mt * 16 + khalf * 8 + r) * Nc + col] = acc[j][r];
    }
}

// ---------------------------------------------------------------------------
// Per-(node, head) attention scores: s = sum_c h[n,hd,c] * a[hd,c]
// ---------------------------------------------------------------------------
__global__ void gat_scores(const float* __restrict__ hlin,
                           const float* __restrict__ a_src, const float* __restrict__ a_dst,
                           float* __restrict__ ssrc, float* __restrict__ sdst,
                           int n, int Hh, int Ch) {
    int idx = blockIdx.x * blockDim.x + threadIdx.x;
    if (idx >= n * Hh) return;
    int node = idx / Hh, hd = idx - node * Hh;
    const float* row = hlin + (size_t)node * Hh * Ch + (size_t)hd * Ch;
    const float* as  = a_src + (size_t)hd * Ch;
    const float* ad  = a_dst + (size_t)hd * Ch;
    float ss = 0.f, sd = 0.f;
    for (int c = 0; c < Ch; ++c) {
        float v = row[c];
        ss += v * as[c];
        sd += v * ad[c];
    }
    ssrc[idx] = ss;
    sdst[idx] = sd;
}

// ---------------------------------------------------------------------------
// Per-(edge, head): leaky_relu score, store it, atomic-max into dst buckets.
// Edges e >= E are self-loops (src = dst = e - E).
// ---------------------------------------------------------------------------
__global__ void edge_scores(const long long* __restrict__ ei, int E_, int n, int Hh,
                            const float* __restrict__ ssrc, const float* __restrict__ sdst,
                            float* __restrict__ ebuf, float* __restrict__ mbuf) {
    int idx = blockIdx.x * blockDim.x + threadIdx.x;
    int total = (E_ + n) * Hh;
    if (idx >= total) return;
    int e = idx / Hh, hd = idx - e * Hh;
    long long s, d;
    if (e < E_) { s = ei[e]; d = ei[(size_t)E_ + e]; }
    else        { s = d = (long long)(e - E_); }
    float v = ssrc[(size_t)s * Hh + hd] + sdst[(size_t)d * Hh + hd];
    v = v > 0.f ? v : 0.2f * v;                  // leaky_relu(0.2)
    ebuf[idx] = v;
    atomicMaxF(&mbuf[(size_t)d * Hh + hd], v);
}

// exp(e - max) in place, atomic sum into z
__global__ void edge_expsum(const long long* __restrict__ ei, int E_, int n, int Hh,
                            float* __restrict__ ebuf,
                            const float* __restrict__ mbuf, float* __restrict__ zbuf) {
    int idx = blockIdx.x * blockDim.x + threadIdx.x;
    int total = (E_ + n) * Hh;
    if (idx >= total) return;
    int e = idx / Hh, hd = idx - e * Hh;
    long long d = (e < E_) ? ei[(size_t)E_ + e] : (long long)(e - E_);
    float x = __expf(ebuf[idx] - mbuf[(size_t)d * Hh + hd]);
    ebuf[idx] = x;
    atomicAdd(&zbuf[(size_t)d * Hh + hd], x);
}

// ---------------------------------------------------------------------------
// One wave per edge: agg[dst] += h[src] * alpha  (HW global_atomic_add_f32,
// 32-lane coalesced row copy; h rows are L2-resident on MI455X's 192MB L2)
// ---------------------------------------------------------------------------
__global__ void gat_scatter(const long long* __restrict__ ei, int E_, int n, int Hh, int Ch,
                            const float* __restrict__ hlin,
                            const float* __restrict__ ebuf, const float* __restrict__ zbuf,
                            float* __restrict__ agg) {
    int gw = (int)((blockIdx.x * (size_t)blockDim.x + threadIdx.x) >> 5);
    int lane = threadIdx.x & 31;
    int Etot = E_ + n;
    if (gw >= Etot) return;
    long long s, d;
    if (gw < E_) { s = ei[gw]; d = ei[(size_t)E_ + gw]; }
    else         { s = d = (long long)(gw - E_); }
    int F = Hh * Ch;
    const float* hs = hlin + (size_t)s * F;
    float*       ag = agg  + (size_t)d * F;
    for (int c = lane; c < F; c += 32) {
        int hd = c / Ch;
        float alpha = ebuf[(size_t)gw * Hh + hd] /
                      (zbuf[(size_t)d * Hh + hd] + 1e-16f);
        atomicAdd(&ag[c], hs[c] * alpha);
    }
}

// ---------------------------------------------------------------------------
// BatchNorm stats on (x + bias): one block = 16 channels x 16 row-lanes, so
// consecutive lanes read consecutive channels (64B contiguous) -> coalesced.
// ---------------------------------------------------------------------------
__global__ void bn_stats(const float* __restrict__ xin, const float* __restrict__ bias,
                         float* __restrict__ meanv, float* __restrict__ varv,
                         int n, int F) {
    int f0 = blockIdx.x * 16;
    int c  = threadIdx.x & 15;
    int r  = threadIdx.x >> 4;
    int f  = f0 + c;
    float b = bias[f];
    float s = 0.f, s2 = 0.f;
    for (int i = r; i < n; i += 16) {
        float v = xin[(size_t)i * F + f] + b;
        s += v; s2 += v * v;
    }
    __shared__ float sh[256], sh2[256];
    sh[threadIdx.x] = s; sh2[threadIdx.x] = s2;
    __syncthreads();
    for (int off = 128; off >= 16; off >>= 1) {
        if (threadIdx.x < off) {
            sh[threadIdx.x]  += sh[threadIdx.x + off];
            sh2[threadIdx.x] += sh2[threadIdx.x + off];
        }
        __syncthreads();
    }
    if (threadIdx.x < 16) {
        float mu = sh[threadIdx.x] / n;
        meanv[f0 + threadIdx.x] = mu;
        varv[f0 + threadIdx.x]  = sh2[threadIdx.x] / n - mu * mu;
    }
}

// BN + ELU apply; writes fp32 output AND bf16 copy (A-matrix for next GEMM)
__global__ void bn_apply(const float* __restrict__ xin, const float* __restrict__ bias,
                         const float* __restrict__ meanv, const float* __restrict__ varv,
                         const float* __restrict__ g, const float* __restrict__ be,
                         float* __restrict__ yout, unsigned short* __restrict__ ybf,
                         int n, int F) {
    size_t total = (size_t)n * F;
    for (size_t i = blockIdx.x * (size_t)blockDim.x + threadIdx.x; i < total;
         i += (size_t)gridDim.x * blockDim.x) {
        int f = (int)(i % F);
        float v = xin[i] + bias[f];
        v = (v - meanv[f]) * rsqrtf(varv[f] + 1e-5f) * g[f] + be[f];
        v = eluf(v);
        yout[i] = v;
        ybf[i]  = f2bf(v);
    }
}

// ---------------------------------------------------------------------------
// Fused head MLP: 32 ->p1-> 64 ->LN->ELU ->p2-> 32 ->LN->ELU ->p3-> 16
// ->ELU ->p4-> 1.  One wave per node (8 nodes / 256-thread block), weights in
// LDS, LayerNorm via wave32 __shfl_xor reductions.
// ---------------------------------------------------------------------------
__global__ void mlp_head(const float* __restrict__ hin,
                         const float* __restrict__ p1w, const float* __restrict__ p1b,
                         const float* __restrict__ ln1g, const float* __restrict__ ln1b,
                         const float* __restrict__ p2w, const float* __restrict__ p2b,
                         const float* __restrict__ ln2g, const float* __restrict__ ln2b,
                         const float* __restrict__ p3w, const float* __restrict__ p3b,
                         const float* __restrict__ p4w, const float* __restrict__ p4b,
                         float* __restrict__ out, int n) {
    __shared__ float W1[32 * 64], W2[64 * 32], W3[32 * 16];
    __shared__ float B1[64], G1[64], E1[64];
    __shared__ float B2[32], G2[32], E2[32];
    __shared__ float B3[16], W4[16];
    __shared__ float stg[8][64];
    __shared__ float h32[8][32];

    int t = threadIdx.x;
    for (int i = t; i < 2048; i += 256) { W1[i] = p1w[i]; W2[i] = p2w[i]; }
    for (int i = t; i < 512;  i += 256) W3[i] = p3w[i];
    if (t < 64) { B1[t] = p1b[t]; G1[t] = ln1g[t]; E1[t] = ln1b[t]; }
    if (t < 32) { B2[t] = p2b[t]; G2[t] = ln2g[t]; E2[t] = ln2b[t]; }
    if (t < 16) { B3[t] = p3b[t]; W4[t] = p4w[t]; }
    __syncthreads();

    int w = t >> 5, lane = t & 31;
    int node = blockIdx.x * 8 + w;
    bool valid = node < n;
    h32[w][lane] = valid ? hin[(size_t)node * 32 + lane] : 0.f;
    __syncthreads();

    // --- p1 (32->64) + LN(64) + ELU : each lane owns 2 outputs ---
    float t0 = B1[lane], t1 = B1[lane + 32];
    for (int k = 0; k < 32; ++k) {
        float hv = h32[w][k];
        t0 += hv * W1[k * 64 + lane];
        t1 += hv * W1[k * 64 + lane + 32];
    }
    float s = t0 + t1, s2 = t0 * t0 + t1 * t1;
    for (int off = 16; off; off >>= 1) { s += __shfl_xor(s, off); s2 += __shfl_xor(s2, off); }
    float mu = s * (1.f / 64.f), var = s2 * (1.f / 64.f) - mu * mu;
    float inv = rsqrtf(var + 1e-5f);
    t0 = eluf((t0 - mu) * inv * G1[lane]      + E1[lane]);
    t1 = eluf((t1 - mu) * inv * G1[lane + 32] + E1[lane + 32]);
    stg[w][lane] = t0; stg[w][lane + 32] = t1;
    __syncthreads();

    // --- p2 (64->32) + LN(32) + ELU : one output per lane ---
    float u = B2[lane];
    for (int k = 0; k < 64; ++k) u += stg[w][k] * W2[k * 32 + lane];
    float su = u, su2 = u * u;
    for (int off = 16; off; off >>= 1) { su += __shfl_xor(su, off); su2 += __shfl_xor(su2, off); }
    mu = su * (1.f / 32.f); var = su2 * (1.f / 32.f) - mu * mu;
    inv = rsqrtf(var + 1e-5f);
    u = eluf((u - mu) * inv * G2[lane] + E2[lane]);
    __syncthreads();
    h32[w][lane] = u;
    __syncthreads();

    // --- p3 (32->16) + ELU ---
    float v3 = 0.f;
    if (lane < 16) {
        v3 = B3[lane];
        for (int k = 0; k < 32; ++k) v3 += h32[w][k] * W3[k * 16 + lane];
        v3 = eluf(v3);
    }
    // --- p4 (16->1) via wave reduction ---
    float contrib = (lane < 16) ? v3 * W4[lane] : 0.f;
    for (int off = 16; off; off >>= 1) contrib += __shfl_xor(contrib, off);
    if (lane == 0 && valid) out[node] = contrib + p4b[0];
}

// ---------------------------------------------------------------------------
// Host orchestration
// ---------------------------------------------------------------------------
extern "C" void kernel_launch(void* const* d_in, const int* in_sizes, int n_in,
                              void* d_out, int out_size, void* d_ws, size_t ws_size,
                              hipStream_t stream) {
    (void)in_sizes; (void)n_in; (void)out_size; (void)ws_size;

    const int N_ = N_NODES, E_ = N_EDGES, Etot_ = E_ + N_;

    // ---- inputs (setup_inputs dict order) ----
    const float*     x    = (const float*)d_in[0];
    const long long* ei   = (const long long*)d_in[1];
    const float *w_in=(const float*)d_in[2],  *b_in=(const float*)d_in[3];
    const float *g_in=(const float*)d_in[4],  *be_in=(const float*)d_in[5];
    const float *w1=(const float*)d_in[6],  *as1=(const float*)d_in[7],  *ad1=(const float*)d_in[8];
    const float *b1=(const float*)d_in[9],  *g1=(const float*)d_in[10], *be1=(const float*)d_in[11];
    const float *w2=(const float*)d_in[12], *as2=(const float*)d_in[13], *ad2=(const float*)d_in[14];
    const float *b2=(const float*)d_in[15], *g2=(const float*)d_in[16], *be2=(const float*)d_in[17];
    const float *w3=(const float*)d_in[18], *as3=(const float*)d_in[19], *ad3=(const float*)d_in[20];
    const float *b3=(const float*)d_in[21], *g3=(const float*)d_in[22], *be3=(const float*)d_in[23];
    const float *wf=(const float*)d_in[24], *asf=(const float*)d_in[25], *adf=(const float*)d_in[26];
    const float *bfb=(const float*)d_in[27],*gf=(const float*)d_in[28], *bef=(const float*)d_in[29];
    const float *p1w=(const float*)d_in[30],*p1b=(const float*)d_in[31];
    const float *ln1g=(const float*)d_in[32],*ln1b=(const float*)d_in[33];
    const float *p2w=(const float*)d_in[34],*p2b=(const float*)d_in[35];
    const float *ln2g=(const float*)d_in[36],*ln2b=(const float*)d_in[37];
    const float *p3w=(const float*)d_in[38],*p3b=(const float*)d_in[39];
    const float *p4w=(const float*)d_in[40],*p4b=(const float*)d_in[41];

    // ---- workspace carve-up (~116 MB) ----
    char* wsb = (char*)d_ws;
    size_t off = 0;
    auto carve = [&](size_t bytes) -> void* {
        void* p = wsb + off;
        off = (off + bytes + 255) & ~(size_t)255;
        return p;
    };
    unsigned short* hbf  = (unsigned short*)carve((size_t)N_ * 512 * 2);  // bf16 GEMM A
    float* hlin  = (float*)carve((size_t)N_ * 512 * 4);   // GEMM out / BN out (ping-pong)
    float* agg   = (float*)carve((size_t)N_ * 512 * 4);   // scatter accumulator
    float* ssrc  = (float*)carve((size_t)N_ * 8 * 4);
    float* sdst  = (float*)carve((size_t)N_ * 8 * 4);
    float* mbuf  = (float*)carve((size_t)N_ * 8 * 4);
    float* zbuf  = (float*)carve((size_t)N_ * 8 * 4);
    float* ebuf  = (float*)carve((size_t)Etot_ * 8 * 4);  // score then exp, in place
    unsigned int* wpack = (unsigned int*)carve((size_t)512 * 512 * 2);
    float* meanv = (float*)carve(512 * 4);
    float* varv  = (float*)carve(512 * 4);

    auto gemm = [&](const float* W, int K, int Nc) {
        int ntiles = Nc >> 4, ktiles = K >> 5;
        int packTotal = ktiles * ntiles * 256;
        pack_b_wmma<<<(packTotal + 255) / 256, 256, 0, stream>>>(W, wpack, K, Nc);
        if ((ntiles & 3) == 0) {
            int waves = (N_ / 16) * (ntiles / 4);
            gemm_wmma_bf16<4><<<(waves * 32 + 255) / 256, 256, 0, stream>>>(
                hbf, wpack, hlin, K, Nc, waves);
        } else {
            int waves = (N_ / 16) * (ntiles / 2);
            gemm_wmma_bf16<2><<<(waves * 32 + 255) / 256, 256, 0, stream>>>(
                hbf, wpack, hlin, K, Nc, waves);
        }
    };

    auto gat_layer = [&](int Fin, int Hh, int Ch, const float* W,
                         const float* as, const float* ad, const float* bias,
                         const float* g, const float* be) {
        int F = Hh * Ch;
        gemm(W, Fin, F);                                             // hlin = h_bf @ W
        int tn = N_ * Hh;
        gat_scores<<<(tn + 255) / 256, 256, 0, stream>>>(hlin, as, ad, ssrc, sdst, N_, Hh, Ch);
        fill_f32<<<512, 256, 0, stream>>>(mbuf, -__builtin_inff(), (size_t)N_ * Hh);
        fill_f32<<<512, 256, 0, stream>>>(zbuf, 0.f, (size_t)N_ * Hh);
        fill_f32<<<2048, 256, 0, stream>>>(agg, 0.f, (size_t)N_ * F);
        int te = Etot_ * Hh;
        edge_scores<<<(te + 255) / 256, 256, 0, stream>>>(ei, E_, N_, Hh, ssrc, sdst, ebuf, mbuf);
        edge_expsum<<<(te + 255) / 256, 256, 0, stream>>>(ei, E_, N_, Hh, ebuf, mbuf, zbuf);
        gat_scatter<<<(Etot_ + 7) / 8, 256, 0, stream>>>(ei, E_, N_, Hh, Ch, hlin, ebuf, zbuf, agg);
        bn_stats<<<F / 16, 256, 0, stream>>>(agg, bias, meanv, varv, N_, F);
        bn_apply<<<2048, 256, 0, stream>>>(agg, bias, meanv, varv, g, be, hlin, hbf, N_, F);
    };

    // ---- input layer: elu(bn(x @ w_in + b_in)) ----
    cast_bf16_k<<<2048, 256, 0, stream>>>(x, hbf, (size_t)N_ * 128);
    gemm(w_in, 128, 64);
    bn_stats<<<64 / 16, 256, 0, stream>>>(hlin, b_in, meanv, varv, N_, 64);
    bn_apply<<<1024, 256, 0, stream>>>(hlin, b_in, meanv, varv, g_in, be_in, hlin, hbf, N_, 64);

    // ---- GAT stack ----
    gat_layer( 64, 8, 64, w1, as1, ad1, b1,  g1, be1);
    gat_layer(512, 8, 64, w2, as2, ad2, b2,  g2, be2);
    gat_layer(512, 4, 32, w3, as3, ad3, b3,  g3, be3);
    gat_layer(128, 1, 32, wf, asf, adf, bfb, gf, bef);

    // ---- fused head MLP -> d_out[N] ----
    mlp_head<<<(N_ + 7) / 8, 256, 0, stream>>>(
        hlin, p1w, p1b, ln1g, ln1b, p2w, p2b, ln2g, ln2b,
        p3w, p3b, p4w, p4b, (float*)d_out, N_);
}
